// LidarYOLOX_39444979646849
// MI455X (gfx1250) — matrix operand order repaired and checked by reading, not compiled, not run on hardware
//
#include <hip/hip_runtime.h>

#define NPTS 30000
#define NEDG 400000

typedef __attribute__((ext_vector_type(16))) __bf16 bf16x16;
typedef __attribute__((ext_vector_type(8)))  __bf16 bf16x8;
typedef __attribute__((ext_vector_type(8)))  float  f32x8;

__device__ __forceinline__ void atomic_max_f32(float* addr, float v) {
    // native float max atomic (no return) — GLOBAL_ATOMIC_MAX_NUM_F32, device scope
    asm volatile("global_atomic_max_num_f32 %0, %1, off scope:SCOPE_DEV"
                 :: "v"(addr), "v"(v) : "memory");
}

// Async global->LDS 16-byte copy (GLOBAL_LOAD_ASYNC_TO_LDS_B128, ASYNCcnt).
// VDST = VGPR holding LDS byte address (flat LDS address low 32 bits == LDS offset),
// VADDR = 64-bit global address.
__device__ __forceinline__ void async_copy16_g2l(void* lds, const void* g) {
    unsigned lo = (unsigned)(unsigned long long)lds;
    asm volatile("global_load_async_to_lds_b128 %0, %1, off"
                 :: "v"(lo), "v"(g) : "memory");
}
__device__ __forceinline__ void wait_async0() {
#if __has_builtin(__builtin_amdgcn_s_wait_asynccnt)
    __builtin_amdgcn_s_wait_asynccnt(0);
#else
    asm volatile("s_wait_asynccnt 0x0" ::: "memory");
#endif
}

__device__ __forceinline__ bf16x16 cat16(bf16x8 lo, bf16x8 hi) {
    return __builtin_shufflevector(lo, hi, 0,1,2,3,4,5,6,7,8,9,10,11,12,13,14,15);
}

// ---------------------------------------------------------------------------
// Pack a fp32 row-major [KSRC x NCOLS] weight matrix into WMMA bf16 B-fragment
// order, zero-padded to KB*32 rows. B layout used here, contiguous K (32x16 tile):
//   lane l: column N = l&15, half h = l>>4, element e: K = 16*h + e
// Fragment storage: frag[((nt*KB + kb)*32 + lane)*16 + e]  (32B/lane, contiguous)
// ---------------------------------------------------------------------------
template<int KSRC, int NCOLS, int KB>
__global__ void pack_b_frag(const float* __restrict__ w, __bf16* __restrict__ out) {
    constexpr int NT = NCOLS / 16;
    int t = blockIdx.x * blockDim.x + threadIdx.x;
    if (t >= NT * KB * 32) return;
    int lane = t & 31;
    int kb   = (t >> 5) % KB;
    int nt   = (t >> 5) / KB;
    int h    = lane >> 4;
    int n    = nt * 16 + (lane & 15);
    __bf16* o = out + (size_t)t * 16;
#pragma unroll
    for (int e = 0; e < 16; ++e) {
        int k = kb * 32 + 16 * h + e;
        float v = (k < KSRC) ? w[k * NCOLS + n] : 0.0f;
        o[e] = (__bf16)v;
    }
}

__global__ void fill_f32_k(float* __restrict__ p, float v, int n) {
    int i = blockIdx.x * blockDim.x + threadIdx.x;
    if (i < n) p[i] = v;
}

__global__ void f32_to_bf16_k(const float* __restrict__ x, __bf16* __restrict__ o, int n) {
    int i = blockIdx.x * blockDim.x + threadIdx.x;
    if (i < n) o[i] = (__bf16)x[i];
}

// where(isfinite(agg), agg, 0) -> bf16 node table for the next layer's gather
__global__ void finitize_bf16_k(const float* __restrict__ agg, __bf16* __restrict__ feat, int n) {
    int i = blockIdx.x * blockDim.x + threadIdx.x;
    if (i < n) {
        float v = agg[i];
        feat[i] = (__bf16)(__builtin_isfinite(v) ? v : 0.0f);
    }
}

// ---------------------------------------------------------------------------
// One PointNet-conv layer over a 128-edge tile:
//   msg = relu(concat(feat[src], pos[src]-pos[dst]) @ Wa + ba) @ Wb + bb
//   agg[dst] = max(agg[dst], msg)      (float atomic max == segment_max)
// CIN   : real input feature width (16 / 64 / 128)
// CINP  : CIN+3 padded to multiple of 32 (32 / 96 / 160)
// COUT  : hidden == output width (64 / 128 / 256)
// Wave w owns M-tile w (A-stationary); streams all N-tiles of packed weights.
// Dual accumulators (even/odd K-step) double WMMA ILP when KB > 1.
// ---------------------------------------------------------------------------
template<int CIN, int CINP, int COUT>
__global__ __launch_bounds__(256)
void edge_layer(const __bf16* __restrict__ feat, const float* __restrict__ pos,
                const int* __restrict__ srcIdx, const int* __restrict__ dstIdx,
                const __bf16* __restrict__ waP, const float* __restrict__ ba,
                const __bf16* __restrict__ wbP, const float* __restrict__ bb,
                float* __restrict__ agg) {
    constexpr int ET  = 128;         // edges per block (8 M-tiles)
    constexpr int KB1 = CINP / 32;   // K-steps GEMM1
    constexpr int KB2 = COUT / 32;   // K-steps GEMM2
    constexpr int NT  = COUT / 16;   // N tiles

    __shared__ __align__(16) __bf16 Ain[ET * CINP];
    __shared__ __align__(16) __bf16 Hid[ET * COUT];
    __shared__ int sSrc[ET], sDst[ET];

    const int tid = threadIdx.x;
    const int e0  = blockIdx.x * ET;
    if (tid < ET) { sSrc[tid] = srcIdx[e0 + tid]; sDst[tid] = dstIdx[e0 + tid]; }
    __syncthreads();

    // ---- gather: feature block via async global->LDS, rel+pad via VALU ----
    constexpr int CH = CIN / 8;
    for (int idx = tid; idx < ET * CH; idx += 256) {
        int e = idx / CH, ch = idx - e * CH;
        async_copy16_g2l(Ain + e * CINP + ch * 8,
                         feat + (size_t)sSrc[e] * CIN + ch * 8);
    }
    constexpr int PADW = CINP - CIN;
    for (int idx = tid; idx < ET * PADW; idx += 256) {
        int e = idx / PADW, c = CIN + (idx - (idx / PADW) * PADW);
        __bf16 v = (__bf16)0.0f;
        if (c < CIN + 3) {
            int cc = c - CIN;
            v = (__bf16)(pos[sSrc[e] * 3 + cc] - pos[sDst[e] * 3 + cc]);
        }
        Ain[e * CINP + c] = v;
    }
    wait_async0();
    __syncthreads();

    const int lane = tid & 31;
    const int wave = tid >> 5;
    const int m    = lane & 15;
    const int h    = lane >> 4;

    // ---- GEMM1: Hid = relu(Ain @ Wa + ba), bf16 into LDS ----
    for (int t = wave; t < 8 * NT; t += 8) {
        int mt = t & 7, nt = t >> 3;
        f32x8 acc0 = {}, acc1 = {};
#pragma unroll
        for (int kb = 0; kb < KB1; ++kb) {
            const __bf16* ap = Ain + (mt * 16 + m) * CINP + kb * 32 + 16 * h;
            bf16x16 a = cat16(*(const bf16x8*)ap, *(const bf16x8*)(ap + 8));
            bf16x16 b = *(const bf16x16*)(waP + ((size_t)(nt * KB1 + kb) * 32 + lane) * 16);
            if (kb & 1)
                acc1 = __builtin_amdgcn_wmma_f32_16x16x32_bf16(false, a, false, b,
                                                               (short)0, acc1, false, false);
            else
                acc0 = __builtin_amdgcn_wmma_f32_16x16x32_bf16(false, a, false, b,
                                                               (short)0, acc0, false, false);
        }
        f32x8 acc;
        if constexpr (KB1 > 1) acc = acc0 + acc1; else acc = acc0;
        float bias = ba[nt * 16 + m];
#pragma unroll
        for (int r = 0; r < 8; ++r) {
            float v = acc[r] + bias;
            v = v > 0.0f ? v : 0.0f;
            Hid[(mt * 16 + r + 8 * h) * COUT + nt * 16 + m] = (__bf16)v;
        }
    }
    __syncthreads();

    // ---- GEMM2 + segment-max scatter ----
    for (int t = wave; t < 8 * NT; t += 8) {
        int mt = t & 7, nt = t >> 3;
        f32x8 acc0 = {}, acc1 = {};
#pragma unroll
        for (int kb = 0; kb < KB2; ++kb) {
            const __bf16* ap = Hid + (mt * 16 + m) * COUT + kb * 32 + 16 * h;
            bf16x16 a = cat16(*(const bf16x8*)ap, *(const bf16x8*)(ap + 8));
            bf16x16 b = *(const bf16x16*)(wbP + ((size_t)(nt * KB2 + kb) * 32 + lane) * 16);
            if (kb & 1)
                acc1 = __builtin_amdgcn_wmma_f32_16x16x32_bf16(false, a, false, b,
                                                               (short)0, acc1, false, false);
            else
                acc0 = __builtin_amdgcn_wmma_f32_16x16x32_bf16(false, a, false, b,
                                                               (short)0, acc0, false, false);
        }
        f32x8 acc;
        if constexpr (KB2 > 1) acc = acc0 + acc1; else acc = acc0;
        int col = nt * 16 + m;
        float bias = bb[col];
#pragma unroll
        for (int r = 0; r < 8; ++r) {
            int el = mt * 16 + r + 8 * h;
            atomic_max_f32(agg + (size_t)sDst[el] * COUT + col, acc[r] + bias);
        }
    }
}

// ---------------------------------------------------------------------------
// Final projective scatter: one block per point, thread = channel (256).
// ---------------------------------------------------------------------------
__global__ __launch_bounds__(256)
void points_to_image_k(const float* __restrict__ agg3, const float* __restrict__ pos,
                       const int* __restrict__ batch, const float* __restrict__ Km,
                       float* __restrict__ out) {
    int i = blockIdx.x;
    int c = threadIdx.x;
    float X = pos[i * 3 + 0], Y = pos[i * 3 + 1], Z = pos[i * 3 + 2];
    int u = (int)((Km[0] * X / Z + Km[2]) * (80.0f / 640.0f));
    int v = (int)((Km[4] * Y / Z + Km[5]) * (60.0f / 480.0f));
    if (u < 0 || u >= 80 || v < 0 || v >= 60) return;
    float val = agg3[(size_t)i * 256 + c];
    if (!__builtin_isfinite(val)) val = 0.0f;
    int b = batch[i];
    atomicAdd(out + (((size_t)b * 256 + c) * 60 + v) * 80 + u, val);
}

extern "C" void kernel_launch(void* const* d_in, const int* in_sizes, int n_in,
                              void* d_out, int out_size, void* d_ws, size_t ws_size,
                              hipStream_t stream) {
    (void)in_sizes; (void)n_in; (void)ws_size;
    const float* x   = (const float*)d_in[0];
    const float* pos = (const float*)d_in[1];
    const float* Km  = (const float*)d_in[2];
    const float* w11 = (const float*)d_in[3];
    const float* b11 = (const float*)d_in[4];
    const float* w12 = (const float*)d_in[5];
    const float* b12 = (const float*)d_in[6];
    const float* w21 = (const float*)d_in[7];
    const float* b21 = (const float*)d_in[8];
    const float* w22 = (const float*)d_in[9];
    const float* b22 = (const float*)d_in[10];
    const float* w31 = (const float*)d_in[11];
    const float* b31 = (const float*)d_in[12];
    const float* w32 = (const float*)d_in[13];
    const float* b32 = (const float*)d_in[14];
    const int* eidx  = (const int*)d_in[15];
    const int* batch = (const int*)d_in[16];
    const int* src = eidx;
    const int* dst = eidx + NEDG;

    // workspace carving (all offsets 256B aligned)
    size_t off = 0;
    char* wsb = (char*)d_ws;
    auto carve = [&](size_t bytes) -> char* {
        char* p = wsb + off;
        off += (bytes + 255) & ~(size_t)255;
        return p;
    };
    __bf16* feat0 = (__bf16*)carve((size_t)NPTS * 16 * 2);
    __bf16* feat1 = (__bf16*)carve((size_t)NPTS * 64 * 2);
    __bf16* feat2 = (__bf16*)carve((size_t)NPTS * 128 * 2);
    float*  agg   = (float*) carve((size_t)NPTS * 256 * 4);
    __bf16* wa1p  = (__bf16*)carve((size_t)4  * 1 * 32 * 16 * 2);
    __bf16* wb1p  = (__bf16*)carve((size_t)4  * 2 * 32 * 16 * 2);
    __bf16* wa2p  = (__bf16*)carve((size_t)8  * 3 * 32 * 16 * 2);
    __bf16* wb2p  = (__bf16*)carve((size_t)8  * 4 * 32 * 16 * 2);
    __bf16* wa3p  = (__bf16*)carve((size_t)16 * 5 * 32 * 16 * 2);
    __bf16* wb3p  = (__bf16*)carve((size_t)16 * 8 * 32 * 16 * 2);

    const float NINF = -__builtin_inff();

    // weight packing + input conversion
    f32_to_bf16_k<<<(NPTS * 16 + 255) / 256, 256, 0, stream>>>(x, feat0, NPTS * 16);
    pack_b_frag<19, 64, 1><<<1,  256, 0, stream>>>(w11, wa1p);   // 128 threads used
    pack_b_frag<64, 64, 2><<<1,  256, 0, stream>>>(w12, wb1p);   // 256
    pack_b_frag<67, 128, 3><<<3, 256, 0, stream>>>(w21, wa2p);   // 768
    pack_b_frag<128,128, 4><<<4, 256, 0, stream>>>(w22, wb2p);   // 1024
    pack_b_frag<131,256, 5><<<10,256, 0, stream>>>(w31, wa3p);   // 2560
    pack_b_frag<256,256, 8><<<16,256, 0, stream>>>(w32, wb3p);   // 4096

    // layer 1: 19 -> 64
    fill_f32_k<<<(NPTS * 64 + 255) / 256, 256, 0, stream>>>(agg, NINF, NPTS * 64);
    edge_layer<16, 32, 64><<<NEDG / 128, 256, 0, stream>>>(
        feat0, pos, src, dst, wa1p, b11, wb1p, b12, agg);
    finitize_bf16_k<<<(NPTS * 64 + 255) / 256, 256, 0, stream>>>(agg, feat1, NPTS * 64);

    // layer 2: 67 -> 128
    fill_f32_k<<<(NPTS * 128 + 255) / 256, 256, 0, stream>>>(agg, NINF, NPTS * 128);
    edge_layer<64, 96, 128><<<NEDG / 128, 256, 0, stream>>>(
        feat1, pos, src, dst, wa2p, b21, wb2p, b22, agg);
    finitize_bf16_k<<<(NPTS * 128 + 255) / 256, 256, 0, stream>>>(agg, feat2, NPTS * 128);

    // layer 3: 131 -> 256
    fill_f32_k<<<(NPTS * 256 + 255) / 256, 256, 0, stream>>>(agg, NINF, NPTS * 256);
    edge_layer<128, 160, 256><<<NEDG / 128, 256, 0, stream>>>(
        feat2, pos, src, dst, wa3p, b31, wb3p, b32, agg);

    // scatter to image
    fill_f32_k<<<(out_size + 255) / 256, 256, 0, stream>>>((float*)d_out, 0.0f, out_size);
    points_to_image_k<<<NPTS, 256, 0, stream>>>(agg, pos, batch, Km, (float*)d_out);
}
